// Embeddings_23922967838804
// MI455X (gfx1250) — matrix operand order repaired and checked by reading, not compiled, not run on hardware
//
#include <hip/hip_runtime.h>
#include <hip/hip_bf16.h>
#include <stdint.h>

#define NEG_SLOPE 0.2f
#define KNN_K 16

typedef __bf16 bf16_t;
typedef __attribute__((ext_vector_type(16))) __bf16 v16bf;
typedef __attribute__((ext_vector_type(8)))  __bf16 v8bf;
typedef __attribute__((ext_vector_type(8)))  float  v8f;

static __device__ __forceinline__ bf16_t f2bf(float x) {
  unsigned u = __float_as_uint(x);
  unsigned r = (u + 0x7FFFu + ((u >> 16) & 1u)) >> 16;
  unsigned short s = (unsigned short)r;
  return __builtin_bit_cast(bf16_t, s);
}

static __device__ __forceinline__ unsigned fkey(float f) {
  unsigned u = __float_as_uint(f);
  return (u & 0x80000000u) ? ~u : (u | 0x80000000u);
}
static __device__ __forceinline__ float kdecode(unsigned k) {
  return __uint_as_float((k & 0x80000000u) ? (k & 0x7FFFFFFFu) : ~k);
}

// ---------------------------------------------------------------------------
// fp32 -> bf16 conversion (weights / activations)
// ---------------------------------------------------------------------------
__global__ void emb_cvt_bf16(const float* __restrict__ s, bf16_t* __restrict__ d, int n) {
  int i = blockIdx.x * 256 + threadIdx.x;
  if (i < n) d[i] = f2bf(s[i]);
}

// ---------------------------------------------------------------------------
// GEMM, bf16 A-path:  Y[R x 128] = A[R x K] * W^T + bias (+ resid)
//   A is pre-converted bf16; channels >= K1 come gathered from X2_16[gidx[row]].
//   Inner loop per K-step: 2 A loads + 8 B loads + 4 v_wmma_f32_16x16x32_bf16.
//   Epilogue dual-writes f32 (Y) and/or bf16 (Y16).
// ---------------------------------------------------------------------------
__global__ __launch_bounds__(256)
void emb_gemm_a16(const bf16_t* __restrict__ X16, int K1,
                  const bf16_t* __restrict__ X2_16, const int* __restrict__ gidx,
                  const bf16_t* __restrict__ W, int K,
                  const float* __restrict__ bias,
                  const float* __restrict__ resid,
                  float* __restrict__ Y, bf16_t* __restrict__ Y16, int R) {
  int wid  = (blockIdx.x * blockDim.x + threadIdx.x) >> 5;
  int lane = threadIdx.x & 31;
  int rt = wid >> 1;
  int ct = wid & 1;
  if (rt * 16 >= R) return;

  int half = lane >> 4;
  int l15  = lane & 15;
  int arow = rt * 16 + l15;
  int g2   = gidx ? gidx[arow] : 0;

  const bf16_t* rowA = X16 + (size_t)arow * K1;
  const bf16_t* rowB = X2_16 ? (X2_16 + (size_t)g2 * 128 - K1) : rowA;

  v8f acc[4] = {};
  int bk = half * 8;

  for (int k0 = 0; k0 < K; k0 += 32) {
    int c0 = k0 + bk;
    int c1 = k0 + 16 + bk;
    v8bf lo = *(const v8bf*)((c0 < K1 ? rowA : rowB) + c0);
    v8bf hi = *(const v8bf*)((c1 < K1 ? rowA : rowB) + c1);
    v16bf a = __builtin_shufflevector(lo, hi, 0,1,2,3,4,5,6,7,8,9,10,11,12,13,14,15);
    int kb = k0 + half * 8;
    #pragma unroll
    for (int sub = 0; sub < 4; ++sub) {
      int o = ct * 64 + sub * 16 + l15;
      v8bf blo = *(const v8bf*)(W + (size_t)o * K + kb);
      v8bf bhi = *(const v8bf*)(W + (size_t)o * K + kb + 16);
      v16bf b = __builtin_shufflevector(blo, bhi, 0,1,2,3,4,5,6,7,8,9,10,11,12,13,14,15);
      acc[sub] = __builtin_amdgcn_wmma_f32_16x16x32_bf16(
          false, a, false, b, (short)0, acc[sub], false, false);
    }
  }

  #pragma unroll
  for (int sub = 0; sub < 4; ++sub) {
    int col = ct * 64 + sub * 16 + l15;
    float bv = bias[col];
    #pragma unroll
    for (int j = 0; j < 8; ++j) {
      int row = rt * 16 + j + half * 8;
      float val = acc[sub][j] + bv;
      if (resid) val += resid[(size_t)row * 128 + col];
      if (Y)   Y[(size_t)row * 128 + col] = val;
      if (Y16) Y16[(size_t)row * 128 + col] = f2bf(val);
    }
  }
}

// ---------------------------------------------------------------------------
// GEMM, f32 A-path with per-row divisor (segment softmax denominator).
// Used only for the small o-projections (R <= 5120). K fixed at 128.
// ---------------------------------------------------------------------------
__global__ __launch_bounds__(256)
void emb_gemm_of32(const float* __restrict__ X, const bf16_t* __restrict__ W,
                   const float* __restrict__ bias, const float* __restrict__ rowdiv,
                   const float* __restrict__ resid,
                   float* __restrict__ Y, bf16_t* __restrict__ Y16, int R) {
  int wid  = (blockIdx.x * blockDim.x + threadIdx.x) >> 5;
  int lane = threadIdx.x & 31;
  int rt = wid >> 1;
  int ct = wid & 1;
  if (rt * 16 >= R) return;

  int half = lane >> 4;
  int l15  = lane & 15;
  int arow = rt * 16 + l15;
  float inv = rowdiv ? (1.0f / fmaxf(rowdiv[arow], 1e-9f)) : 1.0f;
  const float* rowA = X + (size_t)arow * 128;

  v8f acc[4] = {};
  int bk = half * 8;

  for (int k0 = 0; k0 < 128; k0 += 32) {
    v16bf a;
    #pragma unroll
    for (int h = 0; h < 2; ++h) {
      const float* src = rowA + k0 + bk + h * 16;
      float4 f0 = *(const float4*)src;
      float4 f1 = *(const float4*)(src + 4);
      a[h * 8 + 0] = f2bf(f0.x * inv); a[h * 8 + 1] = f2bf(f0.y * inv);
      a[h * 8 + 2] = f2bf(f0.z * inv); a[h * 8 + 3] = f2bf(f0.w * inv);
      a[h * 8 + 4] = f2bf(f1.x * inv); a[h * 8 + 5] = f2bf(f1.y * inv);
      a[h * 8 + 6] = f2bf(f1.z * inv); a[h * 8 + 7] = f2bf(f1.w * inv);
    }
    int kb = k0 + half * 8;
    #pragma unroll
    for (int sub = 0; sub < 4; ++sub) {
      int o = ct * 64 + sub * 16 + l15;
      v8bf blo = *(const v8bf*)(W + (size_t)o * 128 + kb);
      v8bf bhi = *(const v8bf*)(W + (size_t)o * 128 + kb + 16);
      v16bf b = __builtin_shufflevector(blo, bhi, 0,1,2,3,4,5,6,7,8,9,10,11,12,13,14,15);
      acc[sub] = __builtin_amdgcn_wmma_f32_16x16x32_bf16(
          false, a, false, b, (short)0, acc[sub], false, false);
    }
  }

  #pragma unroll
  for (int sub = 0; sub < 4; ++sub) {
    int col = ct * 64 + sub * 16 + l15;
    float bv = bias[col];
    #pragma unroll
    for (int j = 0; j < 8; ++j) {
      int row = rt * 16 + j + half * 8;
      float val = acc[sub][j] + bv;
      if (resid) val += resid[(size_t)row * 128 + col];
      if (Y)   Y[(size_t)row * 128 + col] = val;
      if (Y16) Y16[(size_t)row * 128 + col] = f2bf(val);
    }
  }
}

// ---------------------------------------------------------------------------
// lin 3->128 on (B,3,Ncols)-layout coords
// ---------------------------------------------------------------------------
__global__ void emb_lin3(const float* __restrict__ coords, int Ncols,
                         const float* __restrict__ W, const float* __restrict__ b,
                         float* __restrict__ out, int R) {
  int gid = blockIdx.x * 256 + threadIdx.x;
  if (gid >= R * 128) return;
  int row = gid >> 7, c = gid & 127;
  int bb = row / Ncols, n = row % Ncols;
  const float* cc = coords + (size_t)bb * 3 * Ncols + n;
  float x0 = cc[0], x1 = cc[Ncols], x2 = cc[2 * Ncols];
  out[(size_t)row * 128 + c] =
      fmaf(x0, W[c * 3 + 0], fmaf(x1, W[c * 3 + 1], fmaf(x2, W[c * 3 + 2], b[c])));
}

// ---------------------------------------------------------------------------
// per-channel sum / sumsq partials + finalize to fused affine
// ---------------------------------------------------------------------------
__global__ void emb_stats(const float* __restrict__ h, int R, float* __restrict__ sums) {
  int c = threadIdx.x;
  int r0 = blockIdx.x * 128;
  int r1 = min(r0 + 128, R);
  float s = 0.f, s2 = 0.f;
  for (int r = r0; r < r1; ++r) {
    float v = h[(size_t)r * 128 + c];
    s += v; s2 += v * v;
  }
  atomicAdd(&sums[c], s);
  atomicAdd(&sums[128 + c], s2);
}

__global__ void emb_stats_fin(const float* __restrict__ sums, float invR,
                              const float* __restrict__ g, const float* __restrict__ be,
                              float* __restrict__ scale, float* __restrict__ shift) {
  int c = threadIdx.x;
  float mu  = sums[c] * invR;
  float var = sums[128 + c] * invR - mu * mu;
  float rs  = rsqrtf(var + 1e-5f);
  float sc  = rs * g[c];
  scale[c] = sc;
  shift[c] = be[c] - mu * sc;
}

// ---------------------------------------------------------------------------
// normalize + LeakyReLU, emit bf16 activations for the following GEMM
// ---------------------------------------------------------------------------
__global__ void emb_norm_bf16(const float* __restrict__ h,
                              const float* __restrict__ scale, const float* __restrict__ shift,
                              bf16_t* __restrict__ out, int total) {
  int i = blockIdx.x * 256 + threadIdx.x;
  if (i >= total) return;
  int c = i & 127;
  float v = fmaf(h[i], scale[c], shift[c]);
  v = (v >= 0.0f) ? v : NEG_SLOPE * v;
  out[i] = f2bf(v);
}

// ---------------------------------------------------------------------------
// pos = lin3(xc - nc[idx]); k += pos; v += pos   (wave per point row)
// ---------------------------------------------------------------------------
__global__ void emb_pos_kv(const float* __restrict__ xcoord, int Nx,
                           const float* __restrict__ ncoord, int ncstride,
                           const int* __restrict__ idxp,
                           const float* __restrict__ pw, const float* __restrict__ pb,
                           float* __restrict__ kbuf, float* __restrict__ vbuf, int Rx) {
  int wid = (blockIdx.x * blockDim.x + threadIdx.x) >> 5;
  int lane = threadIdx.x & 31;
  if (wid >= Rx) return;
  int b = wid / Nx, n = wid % Nx;
  int j = idxp[wid];
  const float* xc = xcoord + (size_t)b * 3 * Nx + n;
  const float* nc = ncoord + (size_t)b * 3 * ncstride + j;
  float dx = xc[0] - nc[0];
  float dy = xc[Nx] - nc[ncstride];
  float dz = xc[2 * Nx] - nc[2 * ncstride];
  #pragma unroll
  for (int e = 0; e < 4; ++e) {
    int c = lane * 4 + e;
    float p = fmaf(dx, pw[c * 3 + 0], fmaf(dy, pw[c * 3 + 1], fmaf(dz, pw[c * 3 + 2], pb[c])));
    size_t o = (size_t)wid * 128 + c;
    kbuf[o] += p;
    vbuf[o] += p;
  }
}

// ---------------------------------------------------------------------------
// score[r] = dot(q[node(r)], k[r]) / sqrt(F); segment max via uint atomicMax
// ---------------------------------------------------------------------------
__global__ void emb_score_max(const float* __restrict__ q, const float* __restrict__ kbuf,
                              const int* __restrict__ idxp, int Nx, int Mn,
                              float* __restrict__ score, unsigned* __restrict__ smaxkey,
                              int Rx) {
  int wid = (blockIdx.x * blockDim.x + threadIdx.x) >> 5;
  int lane = threadIdx.x & 31;
  if (wid >= Rx) return;
  int b = wid / Nx;
  int nrow = b * Mn + idxp[wid];
  float s = 0.f;
  #pragma unroll
  for (int e = 0; e < 4; ++e) {
    int c = lane * 4 + e;
    s += q[(size_t)nrow * 128 + c] * kbuf[(size_t)wid * 128 + c];
  }
  #pragma unroll
  for (int off = 16; off; off >>= 1) s += __shfl_xor(s, off, 32);
  s *= 0.08838834764831845f;  // 1/sqrt(128)
  if (lane == 0) {
    score[wid] = s;
    atomicMax(&smaxkey[nrow], fkey(s));
  }
}

// ---------------------------------------------------------------------------
// w = exp(score - smax); num[node] += w*v; den[node] += w
// ---------------------------------------------------------------------------
__global__ void emb_scatter(const float* __restrict__ score, const unsigned* __restrict__ smaxkey,
                            const float* __restrict__ vbuf, const int* __restrict__ idxp,
                            int Nx, int Mn,
                            float* __restrict__ num, float* __restrict__ den, int Rx) {
  int wid = (blockIdx.x * blockDim.x + threadIdx.x) >> 5;
  int lane = threadIdx.x & 31;
  if (wid >= Rx) return;
  int b = wid / Nx;
  int nrow = b * Mn + idxp[wid];
  float smax = kdecode(smaxkey[nrow]);
  if (!__builtin_isfinite(smax)) smax = 0.f;
  float w = __expf(score[wid] - smax);
  #pragma unroll
  for (int e = 0; e < 4; ++e) {
    int c = lane * 4 + e;
    atomicAdd(&num[(size_t)nrow * 128 + c], w * vbuf[(size_t)wid * 128 + c]);
  }
  if (lane == 0) atomicAdd(&den[nrow], w);
}

// ---------------------------------------------------------------------------
// per-node 16-NN by squared distance (branch-free sorted insertion)
// ---------------------------------------------------------------------------
__global__ void emb_knn_topk(const float* __restrict__ ncoord, int M,
                             int* __restrict__ nb, int BM) {
  int t = blockIdx.x * 256 + threadIdx.x;
  if (t >= BM) return;
  int b = t / M, m = t % M;
  const float* base = ncoord + (size_t)b * 3 * M;
  float mx0 = base[m], mx1 = base[M + m], mx2 = base[2 * M + m];
  float bd[KNN_K];
  int   bi[KNN_K];
  #pragma unroll
  for (int s = 0; s < KNN_K; ++s) { bd[s] = 3.4e38f; bi[s] = 0; }
  for (int n = 0; n < M; ++n) {
    float d0 = base[n] - mx0, d1 = base[M + n] - mx1, d2 = base[2 * M + n] - mx2;
    float cd = d0 * d0 + d1 * d1 + d2 * d2;
    int ci = n;
    #pragma unroll
    for (int s = 0; s < KNN_K; ++s) {
      bool lt = cd < bd[s];
      float td = bd[s]; int ti = bi[s];
      bd[s] = lt ? cd : td;  bi[s] = lt ? ci : ti;
      cd    = lt ? td : cd;  ci    = lt ? ti : ci;
    }
  }
  #pragma unroll
  for (int s = 0; s < KNN_K; ++s) nb[(size_t)t * KNN_K + s] = bi[s];
}

// ---------------------------------------------------------------------------
// KNN attention (wave per node row): softmax over 16 neighbors with pos bias
// ---------------------------------------------------------------------------
__global__ void emb_knn_attend(const float* __restrict__ ncoord, int M,
                               const int* __restrict__ nb,
                               const float* __restrict__ q, const float* __restrict__ k,
                               const float* __restrict__ v,
                               const float* __restrict__ pw, const float* __restrict__ pb,
                               float* __restrict__ agg, int BM) {
  int wid = (blockIdx.x * blockDim.x + threadIdx.x) >> 5;
  int lane = threadIdx.x & 31;
  if (wid >= BM) return;
  int b = wid / M, m = wid % M;
  const float* base = ncoord + (size_t)b * 3 * M;
  float mx0 = base[m], mx1 = base[M + m], mx2 = base[2 * M + m];

  float qv[4];
  #pragma unroll
  for (int e = 0; e < 4; ++e) qv[e] = q[(size_t)wid * 128 + lane * 4 + e];

  float sc[KNN_K];
  int   ni[KNN_K];
  #pragma unroll
  for (int j = 0; j < KNN_K; ++j) {
    int n = nb[(size_t)wid * KNN_K + j];
    ni[j] = n;
    int nrow = b * M + n;
    float dx = mx0 - base[n], dy = mx1 - base[M + n], dz = mx2 - base[2 * M + n];
    float s = 0.f;
    #pragma unroll
    for (int e = 0; e < 4; ++e) {
      int c = lane * 4 + e;
      float p = fmaf(dx, pw[c * 3 + 0], fmaf(dy, pw[c * 3 + 1], fmaf(dz, pw[c * 3 + 2], pb[c])));
      s += qv[e] * (k[(size_t)nrow * 128 + c] + p);
    }
    #pragma unroll
    for (int off = 16; off; off >>= 1) s += __shfl_xor(s, off, 32);
    sc[j] = s * 0.08838834764831845f;
  }
  float mxs = sc[0];
  #pragma unroll
  for (int j = 1; j < KNN_K; ++j) mxs = fmaxf(mxs, sc[j]);
  float sum = 0.f;
  #pragma unroll
  for (int j = 0; j < KNN_K; ++j) { sc[j] = __expf(sc[j] - mxs); sum += sc[j]; }
  float isum = 1.0f / sum;

  float acc[4] = {0.f, 0.f, 0.f, 0.f};
  #pragma unroll
  for (int j = 0; j < KNN_K; ++j) {
    int n = ni[j];
    int nrow = b * M + n;
    float aw = sc[j] * isum;
    float dx = mx0 - base[n], dy = mx1 - base[M + n], dz = mx2 - base[2 * M + n];
    #pragma unroll
    for (int e = 0; e < 4; ++e) {
      int c = lane * 4 + e;
      float p = fmaf(dx, pw[c * 3 + 0], fmaf(dy, pw[c * 3 + 1], fmaf(dz, pw[c * 3 + 2], pb[c])));
      acc[e] = fmaf(aw, v[(size_t)nrow * 128 + c] + p, acc[e]);
    }
  }
  #pragma unroll
  for (int e = 0; e < 4; ++e) agg[(size_t)wid * 128 + lane * 4 + e] = acc[e];
}

// ---------------------------------------------------------------------------
// nearest proxy (first-P nodes) per node; writes int index + float copy
// ---------------------------------------------------------------------------
__global__ void emb_proxy_argmin(const float* __restrict__ ncoord, int M, int P,
                                 int* __restrict__ npi, float* __restrict__ npi_f, int BM) {
  int t = blockIdx.x * 256 + threadIdx.x;
  if (t >= BM) return;
  int b = t / M, m = t % M;
  const float* base = ncoord + (size_t)b * 3 * M;
  float mx0 = base[m], mx1 = base[M + m], mx2 = base[2 * M + m];
  float bestd = 3.4e38f;
  int besti = 0;
  for (int p = 0; p < P; ++p) {
    float d0 = base[p] - mx0, d1 = base[M + p] - mx1, d2 = base[2 * M + p] - mx2;
    float d = d0 * d0 + d1 * d1 + d2 * d2;
    if (d < bestd) { bestd = d; besti = p; }
  }
  npi[t] = besti;
  npi_f[t] = (float)besti;
}

// ---------------------------------------------------------------------------
// pf[b,p,:] = nf[b, p, :]  (first P of M node rows), f32 + bf16
// ---------------------------------------------------------------------------
__global__ void emb_copy_pf(const float* __restrict__ nf, float* __restrict__ pf,
                            bf16_t* __restrict__ pf16, int M, int P, int total) {
  int t = blockIdx.x * 256 + threadIdx.x;
  if (t >= total) return;
  int c = t & 127;
  int rp = t >> 7;
  int b = rp / P, p = rp % P;
  float v = nf[((size_t)(b * M + p)) * 128 + c];
  pf[t] = v;
  pf16[t] = f2bf(v);
}

// ===========================================================================
// Host orchestration
// ===========================================================================
extern "C" void kernel_launch(void* const* d_in, const int* in_sizes, int n_in,
                              void* d_out, int out_size, void* d_ws, size_t ws_size,
                              hipStream_t stream) {
  (void)in_sizes; (void)n_in; (void)out_size; (void)ws_size;

  const int Bc = 4, Nn = 40960, Mm = 1280, Fd = 128, Pp = 128;
  const int Rx = Bc * Nn;   // 163840
  const int Rn = Bc * Mm;   // 5120
  const int Rp = Bc * Pp;   // 512

  const float* xin  = (const float*)d_in[0];
  const float* node = (const float*)d_in[1];
  const int*   idx  = (const int*)d_in[2];

  struct LinP { const float *W, *b; };
  struct MlpP { LinP l1; const float *g, *be; LinP l2; };
  struct TfP  { LinP q, k, v, pos, o; };
  int pi = 3;
  auto F32 = [&]() { return (const float*)d_in[pi++]; };
  auto getLin = [&]() { LinP l; l.W = F32(); l.b = F32(); return l; };
  auto getMlp = [&]() { MlpP m; m.l1 = getLin(); m.g = F32(); m.be = F32(); m.l2 = getLin(); return m; };
  auto getTf  = [&]() { TfP t; t.q = getLin(); t.k = getLin(); t.v = getLin(); t.pos = getLin(); t.o = getLin(); return t; };

  MlpP raw = getMlp(), mlp0 = getMlp(), mlp1 = getMlp();
  TfP gt0 = getTf(), gt1 = getTf(), gtN = getTf(), gtP = getTf();
  TfP kn0 = getTf(), kn1 = getTf(), kn2 = getTf();

  // ---- workspace bump allocator ----
  char* wsp = (char*)d_ws;
  size_t used = 0;
  auto alloc = [&](size_t bytes) -> void* {
    size_t a = (used + 255) & ~size_t(255);
    void* p = wsp + a;
    used = a + bytes;
    return p;
  };

  // f32 buffers
  float* kbuf  = (float*)alloc((size_t)Rx * Fd * 4);   // group k / pointnet hidden
  float* vbuf  = (float*)alloc((size_t)Rx * Fd * 4);   // group v
  float* score = (float*)alloc((size_t)Rx * 4);
  float* qn    = (float*)alloc((size_t)Rn * Fd * 4);
  float* knb   = (float*)alloc((size_t)Rn * Fd * 4);   // knn k / raw-node hidden
  float* vnb   = (float*)alloc((size_t)Rn * Fd * 4);
  float* nfA   = (float*)alloc((size_t)Rn * Fd * 4);
  float* nfB   = (float*)alloc((size_t)Rn * Fd * 4);
  float* num   = (float*)alloc((size_t)Rn * Fd * 4);   // also knn agg
  float* den   = (float*)alloc((size_t)Rn * 4);
  unsigned* smaxkey = (unsigned*)alloc((size_t)Rn * 4);
  float* stats = (float*)alloc(256 * 4);
  float* scb   = (float*)alloc(128 * 4);
  float* shb   = (float*)alloc(128 * 4);
  int*   nb    = (int*)alloc((size_t)Rn * KNN_K * 4);
  int*   npi   = (int*)alloc((size_t)Rn * 4);
  float* pf    = (float*)alloc((size_t)Rp * Fd * 4);
  // bf16 activation buffers
  bf16_t* xfA16  = (bf16_t*)alloc((size_t)Rx * Fd * 2);
  bf16_t* xfB16  = (bf16_t*)alloc((size_t)Rx * Fd * 2);
  bf16_t* oxf16  = (bf16_t*)alloc((size_t)Rx * Fd * 2);
  bf16_t* hid16  = (bf16_t*)alloc((size_t)Rx * Fd * 2);
  bf16_t* nfA16  = (bf16_t*)alloc((size_t)Rn * Fd * 2);
  bf16_t* nfB16  = (bf16_t*)alloc((size_t)Rn * Fd * 2);
  bf16_t* onf16  = (bf16_t*)alloc((size_t)Rn * Fd * 2);
  bf16_t* pf16   = (bf16_t*)alloc((size_t)Rp * Fd * 2);

  auto divup = [](int a, int b) { return (a + b - 1) / b; };
  auto cvt = [&](const float* s, int n) -> bf16_t* {
    bf16_t* d = (bf16_t*)alloc((size_t)n * 2);
    emb_cvt_bf16<<<divup(n, 256), 256, 0, stream>>>(s, d, n);
    return d;
  };

  bf16_t* raw_l2w = cvt(raw.l2.W, 128 * 128);
  bf16_t* m0_l1w  = cvt(mlp0.l1.W, 128 * 256);
  bf16_t* m0_l2w  = cvt(mlp0.l2.W, 128 * 128);
  bf16_t* m1_l1w  = cvt(mlp1.l1.W, 128 * 256);
  bf16_t* m1_l2w  = cvt(mlp1.l2.W, 128 * 128);
  struct Tf16 { bf16_t *q, *k, *v, *o; };
  auto mkTf16 = [&](const TfP& t) {
    Tf16 w; w.q = cvt(t.q.W, 16384); w.k = cvt(t.k.W, 16384);
    w.v = cvt(t.v.W, 16384); w.o = cvt(t.o.W, 16384); return w;
  };
  Tf16 gt0w = mkTf16(gt0), gt1w = mkTf16(gt1), gtNw = mkTf16(gtN), gtPw = mkTf16(gtP);
  Tf16 kn0w = mkTf16(kn0), kn1w = mkTf16(kn1), kn2w = mkTf16(kn2);

  auto gemm16 = [&](const bf16_t* X16, int K1, const bf16_t* X2_16, const int* gx,
                    const bf16_t* W, int K, const float* bias, const float* res,
                    float* Y, bf16_t* Y16, int R) {
    emb_gemm_a16<<<dim3(R / 64), dim3(256), 0, stream>>>(
        X16, K1, X2_16, gx, W, K, bias, res, Y, Y16, R);
  };
  auto gemmo = [&](const float* X, const bf16_t* W, const float* bias,
                   const float* rdiv, const float* res, float* Y, bf16_t* Y16, int R) {
    emb_gemm_of32<<<dim3(R / 64), dim3(256), 0, stream>>>(X, W, bias, rdiv, res, Y, Y16, R);
  };

  // ---- mini-pointnet: lin1(f32) -> stats -> norm(bf16) -> l2 GEMM ----
  auto pointnet_norm = [&](float* hid, const MlpP& mp, int R) {
    hipMemsetAsync(stats, 0, 256 * 4, stream);
    emb_stats<<<R / 128, 128, 0, stream>>>(hid, R, stats);
    emb_stats_fin<<<1, 128, 0, stream>>>(stats, 1.0f / (float)R, mp.g, mp.be, scb, shb);
    emb_norm_bf16<<<divup(R * 128, 256), 256, 0, stream>>>(hid, scb, shb, hid16, R * 128);
  };

  // ---- group transformer ----
  auto run_group = [&](const float* xcoord, int Nx, const bf16_t* xfeat16,
                       const float* ncoord, int ncstride, int Mn,
                       const bf16_t* nq16, const float* res_nodes,
                       const int* idxp, const Tf16& w16, const TfP& tp,
                       int Rx_, int Rn_, float* out_nodes, bf16_t* out_nodes16) {
    gemm16(nq16, 128, nullptr, nullptr, w16.q, 128, tp.q.b, nullptr, qn, nullptr, Rn_);
    gemm16(xfeat16, 128, nullptr, nullptr, w16.k, 128, tp.k.b, nullptr, kbuf, nullptr, Rx_);
    gemm16(xfeat16, 128, nullptr, nullptr, w16.v, 128, tp.v.b, nullptr, vbuf, nullptr, Rx_);
    emb_pos_kv<<<Rx_ / 8, 256, 0, stream>>>(xcoord, Nx, ncoord, ncstride, idxp,
                                            tp.pos.W, tp.pos.b, kbuf, vbuf, Rx_);
    hipMemsetAsync(smaxkey, 0, (size_t)Rn_ * 4, stream);
    hipMemsetAsync(den, 0, (size_t)Rn_ * 4, stream);
    hipMemsetAsync(num, 0, (size_t)Rn_ * Fd * 4, stream);
    emb_score_max<<<Rx_ / 8, 256, 0, stream>>>(qn, kbuf, idxp, Nx, Mn, score, smaxkey, Rx_);
    emb_scatter<<<Rx_ / 8, 256, 0, stream>>>(score, smaxkey, vbuf, idxp, Nx, Mn, num, den, Rx_);
    gemmo(num, w16.o, tp.o.b, den, res_nodes, out_nodes, out_nodes16, Rn_);
  };

  float* outf    = (float*)d_out;
  float* out_emb = outf;                 // B*P*F = 65536
  float* out_npi = outf + 65536;         // B*M   = 5120
  float* out_xf  = outf + 70656;         // B*N*F = 20971520
  float* out_nf  = outf + 21042176;      // B*M*F = 655360

  // ---- stage 1: mini-pointnet(raw) on points and nodes ----
  emb_lin3<<<divup(Rx * 128, 256), 256, 0, stream>>>(xin, Nn, raw.l1.W, raw.l1.b, kbuf, Rx);
  pointnet_norm(kbuf, raw, Rx);
  gemm16(hid16, 128, nullptr, nullptr, raw_l2w, 128, raw.l2.b, nullptr, nullptr, xfA16, Rx);

  emb_lin3<<<divup(Rn * 128, 256), 256, 0, stream>>>(node, Mm, raw.l1.W, raw.l1.b, knb, Rn);
  pointnet_norm(knb, raw, Rn);
  gemm16(hid16, 128, nullptr, nullptr, raw_l2w, 128, raw.l2.b, nullptr, nfA, nfA16, Rn);

  // ---- stage 2: gt0 ----
  run_group(xin, Nn, xfA16, node, Mm, Mm, nfA16, nfA, idx, gt0w, gt0, Rx, Rn, nfB, nfB16);

  // ---- stage 3: mlp0 on concat(xfA, gather(nfB, idx)) ----
  gemm16(xfA16, 128, nfB16, idx, m0_l1w, 256, mlp0.l1.b, nullptr, kbuf, nullptr, Rx);
  pointnet_norm(kbuf, mlp0, Rx);
  gemm16(hid16, 128, nullptr, nullptr, m0_l2w, 128, mlp0.l2.b, nullptr, nullptr, xfB16, Rx);

  // ---- stage 4: gt1 ----
  run_group(xin, Nn, xfB16, node, Mm, Mm, nfB16, nfB, idx, gt1w, gt1, Rx, Rn, nfA, nfA16);

  // ---- stage 5: mlp1 -> xf output (f32 to d_out, bf16 for gt_node) ----
  gemm16(xfB16, 128, nfA16, idx, m1_l1w, 256, mlp1.l1.b, nullptr, kbuf, nullptr, Rx);
  pointnet_norm(kbuf, mlp1, Rx);
  gemm16(hid16, 128, nullptr, nullptr, m1_l2w, 128, mlp1.l2.b, nullptr, out_xf, oxf16, Rx);

  // ---- stage 6: gt_node ----
  run_group(xin, Nn, oxf16, node, Mm, Mm, nfA16, nfA, idx, gtNw, gtN, Rx, Rn, nfB, nfB16);

  // ---- stage 7: three KNN transformers on nodes ----
  emb_knn_topk<<<divup(Rn, 256), 256, 0, stream>>>(node, Mm, nb, Rn);
  auto run_knn = [&](const Tf16& w16, const TfP& tp, const bf16_t* nf_in16,
                     const float* nf_in, float* nf_out, bf16_t* nf_out16) {
    gemm16(nf_in16, 128, nullptr, nullptr, w16.q, 128, tp.q.b, nullptr, qn, nullptr, Rn);
    gemm16(nf_in16, 128, nullptr, nullptr, w16.k, 128, tp.k.b, nullptr, knb, nullptr, Rn);
    gemm16(nf_in16, 128, nullptr, nullptr, w16.v, 128, tp.v.b, nullptr, vnb, nullptr, Rn);
    emb_knn_attend<<<Rn / 8, 256, 0, stream>>>(node, Mm, nb, qn, knb, vnb,
                                               tp.pos.W, tp.pos.b, num, Rn);
    gemmo(num, w16.o, tp.o.b, nullptr, nf_in, nf_out, nf_out16, Rn);
  };
  run_knn(kn0w, kn0, nfB16, nfB, nfA, nfA16);
  run_knn(kn1w, kn1, nfA16, nfA, nfB, nfB16);
  run_knn(kn2w, kn2, nfB16, nfB, out_nf, onf16);   // nf output

  // ---- stage 8: proxy attention -> emb ----
  emb_proxy_argmin<<<divup(Rn, 256), 256, 0, stream>>>(node, Mm, Pp, npi, out_npi, Rn);
  emb_copy_pf<<<divup(Rp * 128, 256), 256, 0, stream>>>(out_nf, pf, pf16, Mm, Pp, Rp * 128);
  run_group(node, Mm, onf16, node, Mm, Pp, pf16, pf, npi, gtPw, gtP, Rn, Rp, out_emb, nullptr);
}
